// ChunkGatedDeltaRule_76089640616330
// MI455X (gfx1250) — compile-verified, matrix operand order checked by reading, not run on hardware
//
#include <hip/hip_runtime.h>

// ---------------- problem constants (from reference) ----------------
#define LSEQ 8192
#define NH   8
#define DKD  128
#define DVD  128
#define CH   64              // chunk length
#define NCH  (LSEQ / CH)     // 128 chunks
#define DVSPLIT 4            // independent DV column slices per head
#define DVB  (DVD / DVSPLIT) // 32 columns per block

// padded leading dims (bf16 elements): +8 elems = +16B row skew,
// keeps 16B alignment while rotating LDS banks between rows.
#define LDK (DKD + 8)        // 136 elems -> 272 B rows
#define LDC (CH + 8)         // 72 elems  -> 144 B rows

typedef __attribute__((ext_vector_type(16))) __bf16         bf16x16;
typedef __attribute__((ext_vector_type(16))) unsigned short u16x16;
typedef __attribute__((ext_vector_type(4)))  unsigned int   u32x4;
typedef __attribute__((ext_vector_type(8)))  float          f32x8;

__device__ __forceinline__ unsigned short f2bf(float f) {
  unsigned int u = __builtin_bit_cast(unsigned int, f);
  u += 0x7fffu + ((u >> 16) & 1u);          // round-to-nearest-even
  return (unsigned short)(u >> 16);
}

__device__ __forceinline__ f32x8 wmma_bf(bf16x16 a, bf16x16 b, f32x8 c) {
  // D = A(16x32 bf16) * B(32x16 bf16) + C(16x16 f32)
  return __builtin_amdgcn_wmma_f32_16x16x32_bf16(false, a, false, b,
                                                 (short)0, c, false, false);
}

// Fragment from a row-major [16 x 32] bf16 tile (A operand, or B given a
// transposed-in-LDS source). Two ds_load_b128 per lane (addresses 16B-aligned).
__device__ __forceinline__ bf16x16 frag_row(const unsigned short* p, int ld) {
  const int lane = threadIdx.x & 31;
  const unsigned short* r = p + (lane & 15) * ld + ((lane >> 4) << 3);
  union { u32x4 q[2]; bf16x16 b; } u;
  u.q[0] = *(const u32x4*)(r);
  u.q[1] = *(const u32x4*)(r + 16);
  return u.b;
}

__global__ __launch_bounds__(256, 1)
void gated_delta_chunk_kernel(const float* __restrict__ q,
                              const float* __restrict__ k,
                              const float* __restrict__ v,
                              const float* __restrict__ g,
                              const float* __restrict__ beta,
                              const float* __restrict__ s0,
                              float* __restrict__ out) {
  const int h    = blockIdx.x >> 2;        // head
  const int col0 = (blockIdx.x & 3) * DVB; // DV slice handled by this block
  const int tid  = threadIdx.x;
  const int wave = tid >> 5;
  const int lane = tid & 31;
  const int mo   = (lane >> 4) << 3;       // C/D-layout row offset (0 or 8)
  const int nc   = lane & 15;              // C/D-layout column

  // all WMMA-fed bf16 matrices stored so the K-dim is contiguous in a row
  __shared__ alignas(16) unsigned short SbT[DVB][LDK]; // state^T: [dv][dk]
  __shared__ alignas(16) unsigned short Qb [CH][LDK];  // l2norm(q)*scale
  __shared__ alignas(16) unsigned short Kb [CH][LDK];  // l2norm(k)
  __shared__ alignas(16) unsigned short KdT[DKD][LDC]; // k_j[d]*exp(Gl-Gj)
  __shared__ alignas(16) unsigned short RbT[DVB][LDC]; // RHS^T
  __shared__ alignas(16) unsigned short UbT[DVB][LDC]; // U^T
  __shared__ alignas(16) unsigned short Pb [CH][LDC];  // bf16 T / P
  __shared__ alignas(16) float Vb[CH][DVB];            // V slice (f32)
  __shared__ float Mf[CH][CH];                         // M, reused for P (f32)
  __shared__ float Tf[CH][CH];                         // T = (I+M)^-1 (f32)
  __shared__ float Gc[CH], Eg[CH], El[CH], Bt[CH];

  // persistent fp32 state in accumulators: wave owns DK rows [16w,16w+16)
  f32x8 S[2];
#pragma unroll
  for (int vi = 0; vi < 2; ++vi)
#pragma unroll
    for (int r = 0; r < 8; ++r)
      S[vi][r] = s0[((size_t)h * DKD + wave * 16 + r + mo) * DVD +
                    col0 + vi * 16 + nc];

  const float qscale = 0.08838834764831845f; // 128^-0.5

  for (int c = 0; c < NCH; ++c) {
    const int t0 = c * CH;

    // speculative prefetch of next chunk's q/k/v (global_prefetch_b8)
    if (c + 1 < NCH) {
      const int pr = tid >> 2, ps = tid & 3;
      const size_t pb = ((size_t)(t0 + CH + pr) * NH + h) * DKD + ps * 32;
      __builtin_prefetch(q + pb, 0, 3);
      __builtin_prefetch(k + pb, 0, 3);
      __builtin_prefetch(v + pb, 0, 3);
    }

    // ---- stage 1a: spill state to LDS as bf16 (transposed) ----
#pragma unroll
    for (int vi = 0; vi < 2; ++vi)
#pragma unroll
      for (int r = 0; r < 8; ++r)
        SbT[vi * 16 + nc][wave * 16 + r + mo] = f2bf(S[vi][r]);

    // ---- stage 1b: gates (serial 64-scan is tiny), beta, V slice ----
    if (tid == 0) {
      float acc = 0.f;
      for (int i = 0; i < CH; ++i) {
        acc += g[(size_t)(t0 + i) * NH + h];
        Gc[i] = acc;
        Eg[i] = __expf(acc);
      }
      const float gl = acc;
      for (int i = 0; i < CH; ++i) El[i] = __expf(gl - Gc[i]);
    }
    if (tid < CH) Bt[tid] = beta[(size_t)(t0 + tid) * NH + h];
    // coalesced V staging: 512 float4s over 256 threads
#pragma unroll
    for (int it = 0; it < 2; ++it) {
      const int idx = tid + it * 256;          // [0,512)
      const int row = idx >> 3, c4 = idx & 7;  // 8 float4 per row
      *(float4*)&Vb[row][c4 * 4] =
          *(const float4*)&v[((size_t)(t0 + row) * NH + h) * DVD + col0 + c4 * 4];
    }
    __syncthreads();

    // ---- stage 1c: l2norm q/k (4 lanes per row), fill Qb/Kb/KdT ----
    {
      const int r = tid >> 2, sgm = tid & 3;
      const size_t rowb = ((size_t)(t0 + r) * NH + h) * DKD + sgm * 32;
      const float el = El[r];
      {
        float x[32]; float ss = 0.f;
#pragma unroll
        for (int i4 = 0; i4 < 8; ++i4)
          *(float4*)&x[i4 * 4] = *(const float4*)&q[rowb + i4 * 4];
#pragma unroll
        for (int i = 0; i < 32; ++i) ss += x[i] * x[i];
        ss += __shfl_xor(ss, 1); ss += __shfl_xor(ss, 2);
        const float sc = rsqrtf(ss + 1e-6f) * qscale;
#pragma unroll
        for (int i = 0; i < 32; ++i) Qb[r][sgm * 32 + i] = f2bf(x[i] * sc);
      }
      {
        float x[32]; float ss = 0.f;
#pragma unroll
        for (int i4 = 0; i4 < 8; ++i4)
          *(float4*)&x[i4 * 4] = *(const float4*)&k[rowb + i4 * 4];
#pragma unroll
        for (int i = 0; i < 32; ++i) ss += x[i] * x[i];
        ss += __shfl_xor(ss, 1); ss += __shfl_xor(ss, 2);
        const float sc = rsqrtf(ss + 1e-6f);
#pragma unroll
        for (int i = 0; i < 32; ++i) {
          const float kn = x[i] * sc;
          Kb[r][sgm * 32 + i]  = f2bf(kn);
          KdT[sgm * 32 + i][r] = f2bf(kn * el);   // transpose + decay
        }
      }
    }
    __syncthreads();

    // ---- stage 2: RHS^T = (beta .* (V - Eg .* (K @ S0)))^T ----
    {
      const int mi = wave >> 1, vi = wave & 1;   // 8 tiles, 1 per wave
      f32x8 acc = {0.f,0.f,0.f,0.f,0.f,0.f,0.f,0.f};
#pragma unroll
      for (int kk = 0; kk < 4; ++kk)
        acc = wmma_bf(frag_row(&Kb[mi * 16][kk * 32], LDK),
                      frag_row(&SbT[vi * 16][kk * 32], LDK), acc);
#pragma unroll
      for (int r = 0; r < 8; ++r) {
        const int i = mi * 16 + r + mo, cl = vi * 16 + nc;
        RbT[cl][i] = f2bf(Bt[i] * (Vb[i][cl] - Eg[i] * acc[r]));
      }
    }

    // ---- stage 3: M_ij = beta_i * exp(Gi-Gj) * (K K^T)_ij, j < i ----
#pragma unroll
    for (int t = 0; t < 2; ++t) {
      const int ti = wave * 2 + t, mi = ti >> 2, ni = ti & 3;
      f32x8 acc = {0.f,0.f,0.f,0.f,0.f,0.f,0.f,0.f};
#pragma unroll
      for (int kk = 0; kk < 4; ++kk)
        acc = wmma_bf(frag_row(&Kb[mi * 16][kk * 32], LDK),
                      frag_row(&Kb[ni * 16][kk * 32], LDK), acc); // B = K^T
#pragma unroll
      for (int r = 0; r < 8; ++r) {
        const int i = mi * 16 + r + mo, j = ni * 16 + nc;
        Mf[i][j] = (j < i) ? Bt[i] * __expf(Gc[i] - Gc[j]) * acc[r] : 0.f;
      }
    }
    __syncthreads();

    // ---- stage 4: T = (I+M)^-1, wave-synchronous forward substitution ----
    if (wave == 0) {
      const int c0l = lane, c1l = lane + 32;
      for (int i = 0; i < CH; ++i) {
        float a0 = (i == c0l) ? 1.f : 0.f;
        float a1 = (i == c1l) ? 1.f : 0.f;
        for (int j = 0; j < i; ++j) {
          const float m = Mf[i][j];
          a0 -= m * Tf[j][c0l];
          a1 -= m * Tf[j][c1l];
        }
        Tf[i][c0l] = a0; Tf[i][c1l] = a1;
      }
    }
    __syncthreads();
    for (int idx = tid; idx < CH * CH; idx += 256)
      Pb[idx >> 6][idx & 63] = f2bf((&Tf[0][0])[idx]);
    __syncthreads();

    // ---- stage 5: U^T = (T @ RHS)^T ----
    {
      const int mi = wave >> 1, vi = wave & 1;
      f32x8 acc = {0.f,0.f,0.f,0.f,0.f,0.f,0.f,0.f};
#pragma unroll
      for (int kk = 0; kk < 2; ++kk)
        acc = wmma_bf(frag_row(&Pb[mi * 16][kk * 32], LDC),
                      frag_row(&RbT[vi * 16][kk * 32], LDC), acc);
#pragma unroll
      for (int r = 0; r < 8; ++r)
        UbT[vi * 16 + nc][mi * 16 + r + mo] = f2bf(acc[r]);
    }
    __syncthreads();

    // ---- stage 6: P_ij = exp(Gi-Gj) * (Q K^T)_ij, j <= i (reuse Mf) ----
#pragma unroll
    for (int t = 0; t < 2; ++t) {
      const int ti = wave * 2 + t, mi = ti >> 2, ni = ti & 3;
      f32x8 acc = {0.f,0.f,0.f,0.f,0.f,0.f,0.f,0.f};
#pragma unroll
      for (int kk = 0; kk < 4; ++kk)
        acc = wmma_bf(frag_row(&Qb[mi * 16][kk * 32], LDK),
                      frag_row(&Kb[ni * 16][kk * 32], LDK), acc);
#pragma unroll
      for (int r = 0; r < 8; ++r) {
        const int i = mi * 16 + r + mo, j = ni * 16 + nc;
        Mf[i][j] = (j <= i) ? __expf(Gc[i] - Gc[j]) * acc[r] : 0.f;
      }
    }
    __syncthreads();
    for (int idx = tid; idx < CH * CH; idx += 256)
      Pb[idx >> 6][idx & 63] = f2bf((&Mf[0][0])[idx]);
    __syncthreads();

    // ---- stage 7: O = rowscale(Q @ S0, Eg) + P @ U ----
    {
      const int mi = wave >> 1, vi = wave & 1;
      f32x8 acc = {0.f,0.f,0.f,0.f,0.f,0.f,0.f,0.f};
#pragma unroll
      for (int kk = 0; kk < 4; ++kk)
        acc = wmma_bf(frag_row(&Qb[mi * 16][kk * 32], LDK),
                      frag_row(&SbT[vi * 16][kk * 32], LDK), acc);
#pragma unroll
      for (int r = 0; r < 8; ++r) acc[r] *= Eg[mi * 16 + r + mo];
#pragma unroll
      for (int kk = 0; kk < 2; ++kk)
        acc = wmma_bf(frag_row(&Pb[mi * 16][kk * 32], LDC),
                      frag_row(&UbT[vi * 16][kk * 32], LDC), acc);
#pragma unroll
      for (int r = 0; r < 8; ++r)
        out[((size_t)(t0 + mi * 16 + r + mo) * NH + h) * DVD +
            col0 + vi * 16 + nc] = acc[r];
    }

    // ---- stage 8: S = exp(G_last) * S + KdT @ U (state in accumulators) ----
    {
      const float egl = Eg[CH - 1];
#pragma unroll
      for (int vi = 0; vi < 2; ++vi) {
#pragma unroll
        for (int r = 0; r < 8; ++r) S[vi][r] *= egl;
#pragma unroll
        for (int kk = 0; kk < 2; ++kk)
          S[vi] = wmma_bf(frag_row(&KdT[wave * 16][kk * 32], LDC),
                          frag_row(&UbT[vi * 16][kk * 32], LDC), S[vi]);
      }
    }
    __syncthreads();  // protects SbT/Kb/UbT/... rewrite next chunk
  }

  // ---- final state ----
  float* fs = out + (size_t)LSEQ * NH * DVD;
#pragma unroll
  for (int vi = 0; vi < 2; ++vi)
#pragma unroll
    for (int r = 0; r < 8; ++r)
      fs[((size_t)h * DKD + wave * 16 + r + mo) * DVD + col0 + vi * 16 + nc] =
          S[vi][r];
}

extern "C" void kernel_launch(void* const* d_in, const int* in_sizes, int n_in,
                              void* d_out, int out_size, void* d_ws, size_t ws_size,
                              hipStream_t stream) {
  (void)in_sizes; (void)n_in; (void)out_size; (void)d_ws; (void)ws_size;
  const float* q    = (const float*)d_in[0];
  const float* k    = (const float*)d_in[1];
  const float* v    = (const float*)d_in[2];
  const float* g    = (const float*)d_in[3];
  const float* beta = (const float*)d_in[4];
  const float* s0   = (const float*)d_in[5];
  float* out = (float*)d_out;

  dim3 grid(NH * DVSPLIT);  // 8 heads x 4 DV slices = 32 blocks
  dim3 block(256);          // 8 wave32s
  gated_delta_chunk_kernel<<<grid, block, 0, stream>>>(q, k, v, g, beta, s0, out);
}
